// Quantizing_91001767067775
// MI455X (gfx1250) — compile-verified, matrix operand order checked by reading, not run on hardware
//
#include <hip/hip_runtime.h>
#include <hip/hip_bf16.h>

typedef __attribute__((ext_vector_type(16))) __bf16          v16bf;
typedef __attribute__((ext_vector_type(8)))  float           v8f;
typedef __attribute__((ext_vector_type(8)))  unsigned short  us8;
typedef __attribute__((ext_vector_type(16))) unsigned short  us16;
typedef __attribute__((ext_vector_type(4)))  unsigned int    u32x4;
typedef __attribute__((ext_vector_type(8)))  unsigned int    u32x8;

#define EDIM   256
#define QTOT   4096
#define BM     128        // rows per block
#define BN     128        // codes per chunk
#define WM     16         // rows per wave
#define NCHUNK (QTOT / BN)
#define SLOTS  4096       // 16B fragment slots per chunk (64 KB)

// round-to-nearest-even fp32 -> bf16
__device__ __forceinline__ unsigned short f2bf(float f) {
  unsigned int u = __float_as_uint(f);
  u += 0x7FFFu + ((u >> 16) & 1u);
  return (unsigned short)(u >> 16);
}

// ---------------------------------------------------------------- w2 prepass
__global__ void vq_w2_kernel(const float* __restrict__ W, float* __restrict__ w2) {
  int q = blockIdx.x * blockDim.x + threadIdx.x;
  if (q >= QTOT) return;
  const float4* row = (const float4*)(W + (size_t)q * EDIM);
  float s = 0.f;
#pragma unroll 8
  for (int i = 0; i < EDIM / 4; ++i) {
    float4 v = row[i];
    s += v.x * v.x + v.y * v.y + v.z * v.z + v.w * v.w;
  }
  w2[q] = s;
}

// ------------------------------------------- W -> bf16 fragment-layout prepass
// One 64 KB chunk per block: slot = ((p*8+kk)*8+t)*32 + lane, 16 bytes each.
__global__ void vq_wfrag_kernel(const float* __restrict__ W, us8* __restrict__ Wf) {
  const int chunk = blockIdx.x;
  const int tid   = threadIdx.x;
  const int qbase = chunk * BN;
  us8* dst = Wf + (size_t)chunk * SLOTS;
#pragma unroll
  for (int i = 0; i < 16; ++i) {
    int idx = tid + 256 * i;            // 0..4095
    int L   = idx & 31;
    int t   = (idx >> 5) & 7;
    int kk  = (idx >> 8) & 7;
    int p   = (idx >> 11) & 1;
    int q   = qbase + t * 16 + (L & 15);
    int K0  = 32 * kk + ((L < 16) ? 0 : 8) + 16 * p;
    const float4* src = (const float4*)(W + (size_t)q * EDIM + K0);
    float4 s0 = src[0], s1 = src[1];
    us8 v;
    v[0] = f2bf(s0.x); v[1] = f2bf(s0.y); v[2] = f2bf(s0.z); v[3] = f2bf(s0.w);
    v[4] = f2bf(s1.x); v[5] = f2bf(s1.y); v[6] = f2bf(s1.z); v[7] = f2bf(s1.w);
    dst[((p * 8 + kk) * 8 + t) * 32 + L] = v;
  }
}

// -------------------------------------------------------- TDM chunk DMA issue
// 1-row tensor: data_size=4B, tile_dim0 = 16384 elements = 64 KB, contiguous.
__device__ __forceinline__ void tdm_load_chunk(const void* gsrc, unsigned lds_off) {
  unsigned long long ga = (unsigned long long)(size_t)gsrc;
  u32x4 g0;
  g0[0] = 1u;                                             // count=1 (valid), user mode
  g0[1] = lds_off;                                        // lds_addr (bytes)
  g0[2] = (unsigned)(ga & 0xFFFFFFFFu);                   // global_addr[31:0]
  g0[3] = (unsigned)((ga >> 32) & 0x1FFFFFFu)             // global_addr[56:32]
          | (2u << 30);                                   // type=2 ("image")
  u32x8 g1;
  g1[0] = (2u << 16);                                     // data_size=4B; mask=0; no flags
  g1[1] = (16384u & 0xFFFFu) << 16;                       // tensor_dim0[15:0]
  g1[2] = (16384u >> 16) | (1u << 16);                    // tensor_dim0[31:16] | tensor_dim1=1
  g1[3] = (16384u << 16);                                 // tile_dim0 = 16384
  g1[4] = 1u;                                             // tile_dim1=1, tile_dim2=0
  g1[5] = 16384u;                                         // tensor_dim0_stride[31:0]
  g1[6] = 0u;                                             // stride hi / dim1_stride lo
  g1[7] = 0u;
  asm volatile("tensor_load_to_lds %0, %1" :: "s"(g0), "s"(g1) : "memory");
}

// ---------------------------------------------------------------- main kernel
__global__ __launch_bounds__(256) void vq_argmin_kernel(
    const float* __restrict__ X, const float* __restrict__ W,
    const float* __restrict__ w2, const us8* __restrict__ Wf,
    float* __restrict__ outData, float* __restrict__ outIdx) {
  // Double-buffered B fragments: [buf][plane][kstep][ntile][lane] = 2 x 64 KB
  __shared__ us8 Wbuf[2][2][8][8][32];

  const int tid  = threadIdx.x;
  const int lane = tid & 31;
  const int wv   = tid >> 5;       // wave 0..7
  const int ln16 = lane & 15;
  const int hi   = lane >> 4;      // 0 or 1
  const int h    = hi * 8;         // K half offset inside 32-block

  const int rowBase = blockIdx.x * BM;
  const int myRow   = rowBase + wv * WM + ln16;   // A-matrix M = lane%16
  const float* xrow = X + (size_t)myRow * EDIM;

  // Kick off DMA of chunk 0 while we load A fragments.
  if (wv == 0)
    tdm_load_chunk(Wf, (unsigned)(size_t)&Wbuf[0][0][0][0][0]);

  // ---- A fragments for this wave's 16x256 slab, resident in registers ----
  v16bf afrag[8];
#pragma unroll
  for (int kk = 0; kk < 8; ++kk) {
    const float* p0 = xrow + 32 * kk + h;
    us16 u;
#pragma unroll
    for (int e = 0; e < 8; ++e) {
      u[e]     = f2bf(p0[e]);        // K = 32kk + h + e
      u[e + 8] = f2bf(p0[16 + e]);   // K = 32kk + h + 16 + e
    }
    afrag[kk] = __builtin_bit_cast(v16bf, u);
  }

  float best[8];
  int   bidx[8];
#pragma unroll
  for (int r = 0; r < 8; ++r) { best[r] = 3.4e38f; bidx[r] = 0; }

  const v8f vzero = {0.f, 0.f, 0.f, 0.f, 0.f, 0.f, 0.f, 0.f};

  for (int c = 0; c < NCHUNK; ++c) {
    const int b = c & 1;
    __syncthreads();                       // everyone done with buf[b] (iter c-2)
    if (wv == 0) __builtin_amdgcn_s_wait_tensorcnt(0);   // chunk c landed
    __syncthreads();                       // publish buf[b] to all waves
    if (wv == 0 && c + 1 < NCHUNK)         // overlap DMA of c+1 with compute of c
      tdm_load_chunk(Wf + (size_t)(c + 1) * SLOTS,
                     (unsigned)(size_t)&Wbuf[1 - b][0][0][0][0]);

    // ---- 16x128 tile of scores: 64 WMMAs, fp32 accumulate ----
    v8f acc[8];
#pragma unroll
    for (int t = 0; t < 8; ++t) acc[t] = vzero;

#pragma unroll
    for (int kk = 0; kk < 8; ++kk) {
#pragma unroll
      for (int t = 0; t < 8; ++t) {
        us8  lo  = Wbuf[b][0][kk][t][lane];
        us8  hi8 = Wbuf[b][1][kk][t][lane];
        us16 u   = __builtin_shufflevector(lo, hi8, 0, 1, 2, 3, 4, 5, 6, 7,
                                           8, 9, 10, 11, 12, 13, 14, 15);
        v16bf bm = __builtin_bit_cast(v16bf, u);
        acc[t] = __builtin_amdgcn_wmma_f32_16x16x32_bf16(
            false, afrag[kk], false, bm, (short)0, acc[t], false, false);
      }
    }

    // ---- score = w2[q] - 2*(x.w); running argmin ----
    const int qbase = c * BN;
#pragma unroll
    for (int t = 0; t < 8; ++t) {
      int   q   = qbase + t * 16 + ln16;   // C/D: N = lane%16
      float w2q = w2[q];
#pragma unroll
      for (int r = 0; r < 8; ++r) {        // C/D: M = r + 8*hi
        float s = __builtin_fmaf(-2.f, acc[t][r], w2q);
        if (s < best[r]) { best[r] = s; bidx[r] = q; }
      }
    }
  }

  // ---- reduce across the 16 lanes of each half (rows r and r+8) ----
#pragma unroll
  for (int r = 0; r < 8; ++r) {
#pragma unroll
    for (int m = 8; m >= 1; m >>= 1) {
      float ob = __shfl_xor(best[r], m, 32);
      int   oi = __shfl_xor(bidx[r], m, 32);
      if (ob < best[r] || (ob == best[r] && oi < bidx[r])) {
        best[r] = ob; bidx[r] = oi;
      }
    }
  }

  __syncthreads();  // all waves done with Wbuf -> reuse LDS for indices
  int* bestIdx = (int*)&Wbuf[0][0][0][0][0];
  if (ln16 == 0) {  // lanes 0 (rows r) and 16 (rows r+8) hold winners
#pragma unroll
    for (int r = 0; r < 8; ++r) {
      int row = wv * WM + hi * 8 + r;
      bestIdx[row] = bidx[r];
      outIdx[rowBase + row] = (float)bidx[r];
    }
  }
  __syncthreads();

  // ---- gather winning codebook rows (fp32) to output ----
  {
    int row  = tid >> 1;     // 0..127
    int part = tid & 1;      // half-row of 128 floats
    int qsel = bestIdx[row];
    const float4* src = (const float4*)(W + (size_t)qsel * EDIM) + part * 32;
    float4*       dst = (float4*)(outData + (size_t)(rowBase + row) * EDIM) + part * 32;
#pragma unroll
    for (int i = 0; i < 32; ++i) dst[i] = src[i];
  }
}

// ---------------------------------------------------------------- launcher
extern "C" void kernel_launch(void* const* d_in, const int* in_sizes, int n_in,
                              void* d_out, int out_size, void* d_ws, size_t ws_size,
                              hipStream_t stream) {
  (void)n_in; (void)out_size; (void)ws_size;
  const float* x = (const float*)d_in[0];   // (16,2048,256) f32
  const float* w = (const float*)d_in[1];   // (4096,256) f32

  float* w2 = (float*)d_ws;                           // 16 KB
  us8*   wf = (us8*)((char*)d_ws + QTOT * sizeof(float));  // 2 MB fragment-layout bf16 W

  float* out     = (float*)d_out;
  int    N       = in_sizes[0] / EDIM;      // 32768 rows
  float* outData = out;                     // N*E floats
  float* outIdx  = out + (size_t)N * EDIM;  // N floats (indices as f32)

  vq_w2_kernel<<<(QTOT + 255) / 256, 256, 0, stream>>>(w, w2);
  vq_wfrag_kernel<<<NCHUNK, 256, 0, stream>>>(w, wf);
  vq_argmin_kernel<<<N / BM, 256, 0, stream>>>(x, w, w2, wf, outData, outIdx);
}
